// CascadeRoIHeads_14559939133473
// MI455X (gfx1250) — matrix operand order repaired and checked by reading, not compiled, not run on hardware
//
#include <hip/hip_runtime.h>

// ---------------- constants ----------------
#define POOLSZ    7
#define FEAT_CH   256
#define IN_DIM    (FEAT_CH * POOLSZ * POOLSZ)   // 12544
#define HID       1024
#define NCLS      81
#define NCLS_PAD  96
#define NPROP     512
#define NBATCH    2
#define NROWS     (NBATCH * NPROP)              // 1024
#define DETS      100
#define CLIP_VAL  4.135166556742356f            // log(1000/16)

typedef __attribute__((ext_vector_type(16))) __bf16 v16bf;
typedef __attribute__((ext_vector_type(8)))  __bf16 v8bf;
typedef __attribute__((ext_vector_type(8)))  float  v8f;

union ABu { v16bf v; v8bf h[2]; };

__device__ __forceinline__ unsigned short f2bf(float f) {
  unsigned int u = __float_as_uint(f);
  unsigned int r = u + 0x7FFFu + ((u >> 16) & 1u);
  return (unsigned short)(r >> 16);
}
__device__ __forceinline__ float bf2f(unsigned short h) {
  return __uint_as_float(((unsigned int)h) << 16);
}

// async 16-byte global -> LDS copy (CDNA5 GLOBAL_LOAD_ASYNC_TO_LDS_B128,
// tracked by ASYNCcnt).
__device__ __forceinline__ void async_g2l_b128(void* lds_dst, const void* gsrc) {
  unsigned int  l = (unsigned int)(__SIZE_TYPE__)lds_dst;   // LDS byte offset
  unsigned long long g = (unsigned long long)(__SIZE_TYPE__)gsrc;
  asm volatile("global_load_async_to_lds_b128 %0, %1, off"
               :: "v"(l), "v"(g) : "memory");
}
__device__ __forceinline__ void wait_async_le1() {
  asm volatile("s_wait_asynccnt 0x1" ::: "memory");
}
__device__ __forceinline__ void wait_async_le0() {
  asm volatile("s_wait_asynccnt 0x0" ::: "memory");
}

// ---------------- RoI align -> bf16 activation row ----------------
__global__ void roi_align_k(const float* __restrict__ feat,
                            const float* __restrict__ boxes,
                            unsigned short* __restrict__ Aout) {
  int roi = blockIdx.x;            // 0..NROWS-1
  int b   = roi / NPROP;
  const float* bx = boxes + (size_t)roi * 4;
  __shared__ float sdx[49], sdy[49];
  __shared__ int   sx0[49], sy0[49], sx1[49], sy1[49];
  int tid = threadIdx.x;
  if (tid < 49) {
    float x1 = bx[0] * 0.125f, y1 = bx[1] * 0.125f;
    float x2 = bx[2] * 0.125f, y2 = bx[3] * 0.125f;
    float bw = fmaxf(x2 - x1, 1.0f) * (1.0f / POOLSZ);
    float bh = fmaxf(y2 - y1, 1.0f) * (1.0f / POOLSZ);
    int pi = tid / POOLSZ, pj = tid % POOLSZ;
    float gx = x1 + (pj + 0.5f) * bw;
    float gy = y1 + (pi + 0.5f) * bh;
    float x = fminf(fmaxf(gx, 0.f), 63.f);
    float y = fminf(fmaxf(gy, 0.f), 63.f);
    float x0f = floorf(x), y0f = floorf(y);
    int x0 = (int)x0f, y0 = (int)y0f;
    sx0[tid] = x0; sy0[tid] = y0;
    sx1[tid] = min(x0 + 1, 63); sy1[tid] = min(y0 + 1, 63);
    sdx[tid] = x - x0f; sdy[tid] = y - y0f;
  }
  __syncthreads();
  const float* fb = feat + (size_t)b * FEAT_CH * 64 * 64;
  unsigned short* orow = Aout + (size_t)roi * IN_DIM;
  for (int c = tid; c < FEAT_CH; c += blockDim.x) {
    const float* fc = fb + (size_t)c * 4096;
    for (int p = 0; p < 49; ++p) {
      float dx = sdx[p], dy = sdy[p];
      float v00 = fc[sy0[p] * 64 + sx0[p]];
      float v01 = fc[sy0[p] * 64 + sx1[p]];
      float v10 = fc[sy1[p] * 64 + sx0[p]];
      float v11 = fc[sy1[p] * 64 + sx1[p]];
      float v = v00 * (1.f - dy) * (1.f - dx) + v01 * (1.f - dy) * dx +
                v10 * dy * (1.f - dx)        + v11 * dy * dx;
      orow[c * 49 + p] = f2bf(v);
    }
  }
}

// ---------------- weight pack: f32 (K x srcN) -> bf16 B-fragment tiles ----
// Bp[((nt*KT + kt)*32 + lane)*16 + i] = W[kt*32 + ((lane>=16)?8:0) + (i&7) + ((i>>3)<<4)][nt*16 + (lane&15)]
__global__ void pack_b_k(const float* __restrict__ W,
                         unsigned short* __restrict__ Bp,
                         int K, int N, int srcN) {
  size_t idx = (size_t)blockIdx.x * blockDim.x + threadIdx.x;
  size_t total = (size_t)K * N;
  if (idx >= total) return;
  int i = (int)(idx & 15);
  int l = (int)((idx >> 4) & 31);
  size_t rest = idx >> 9;
  int KT = K >> 5;
  int kt = (int)(rest % KT);
  int nt = (int)(rest / KT);
  int n = nt * 16 + (l & 15);
  int k = kt * 32 + ((l >> 4) << 3) + (i & 7) + ((i >> 3) << 4);
  float v = (n < srcN) ? W[(size_t)k * srcN + n] : 0.f;
  Bp[idx] = f2bf(v);
}

// ---------------- tiled WMMA GEMM with async-LDS A staging ----------------
// Block = 256 thr (8 waves) computes 64(M) x 128(N). Wave (mi,ng) computes a
// 16x64 strip: 1 A fragment reused across 4 WMMAs. A tile (64x32 bf16, 4KB)
// is double-buffered in LDS via GLOBAL_LOAD_ASYNC_TO_LDS_B128 + ASYNCcnt.
// Requires M%64==0, N%128==0, K%32==0. Epilogue: bias + ReLU, bf16 out.
__global__ void wmma_gemm_tiled_k(const unsigned short* __restrict__ A,
                                  const unsigned short* __restrict__ Bp,
                                  const float* __restrict__ bias,
                                  unsigned short* __restrict__ out,
                                  int M, int K, int N) {
  __shared__ unsigned short lds[2][64 * 32];    // 2 x 4KB A tiles

  int nb   = N >> 7;                    // blocks along N
  int bm   = (blockIdx.x / nb) * 64;
  int bn   = (blockIdx.x % nb) * 128;
  int tid  = threadIdx.x;
  int lane = tid & 31;
  int w    = tid >> 5;                  // wave 0..7
  int mi   = w & 3;                     // m-tile within block
  int ng   = w >> 2;                    // n-group (0..1), 64 cols each
  int KT   = K >> 5;

  // cooperative A-stage addressing: 256 thr x 16B = 4KB tile
  int arow_l = tid >> 2;                // 0..63
  int acol_l = (tid & 3) << 3;          // 0,8,16,24 (elements)
  const unsigned short* aglob = A + (size_t)(bm + arow_l) * K + acol_l;
  int lds_off = arow_l * 32 + acol_l;   // element offset within a buffer

  // per-lane A fragment source (ISA 16-bit A 16x32 layout)
  int fr  = mi * 16 + (lane & 15);      // row within 64-row tile
  int fko = (lane >> 4) << 3;           // 0 or 8

  // packed-B pointers: 4 consecutive 16-col tiles for this n-group
  int nt0 = (bn >> 4) + ng * 4;
  const unsigned short* bptr = Bp + ((size_t)nt0 * KT * 32 + lane) * 16;
  size_t btile = (size_t)KT * 512;      // elements between packed n-tiles

  v8f accs[4] = {};

  // prologue: stage first A tile into buffer 0
  async_g2l_b128(&lds[0][lds_off], aglob);
  aglob += 32;

  for (int kt = 0; kt < KT; ++kt) {
    int buf  = kt & 1;
    bool more = (kt + 1) < KT;
    if (more) {                          // stage next tile into other buffer
      async_g2l_b128(&lds[buf ^ 1][lds_off], aglob);
      aglob += 32;
      wait_async_le1();                  // our tile for `buf` is complete
    } else {
      wait_async_le0();
    }
    __syncthreads();                     // whole tile visible to all waves

    ABu a;
    a.h[0] = *(const v8bf*)(&lds[buf][fr * 32 + fko]);       // ds_load_b128
    a.h[1] = *(const v8bf*)(&lds[buf][fr * 32 + fko + 16]);  // ds_load_b128

    v16bf b0 = *(const v16bf*)(bptr);
    v16bf b1 = *(const v16bf*)(bptr + btile);
    v16bf b2 = *(const v16bf*)(bptr + 2 * btile);
    v16bf b3 = *(const v16bf*)(bptr + 3 * btile);
    __builtin_prefetch(bptr + 512, 0, 0);

    accs[0] = __builtin_amdgcn_wmma_f32_16x16x32_bf16(false, a.v, false, b0,
                                                      (short)0, accs[0], false, false);
    accs[1] = __builtin_amdgcn_wmma_f32_16x16x32_bf16(false, a.v, false, b1,
                                                      (short)0, accs[1], false, false);
    accs[2] = __builtin_amdgcn_wmma_f32_16x16x32_bf16(false, a.v, false, b2,
                                                      (short)0, accs[2], false, false);
    accs[3] = __builtin_amdgcn_wmma_f32_16x16x32_bf16(false, a.v, false, b3,
                                                      (short)0, accs[3], false, false);
    bptr += 512;
    __syncthreads();                     // protect lds[buf] from next stage
  }

  int rbase = bm + mi * 16 + ((lane >> 4) << 3);
  int cbase = lane & 15;
#pragma unroll
  for (int j = 0; j < 4; ++j) {
    int col = (nt0 + j) * 16 + cbase;
    float bcol = bias[col];
#pragma unroll
    for (int r = 0; r < 8; ++r) {
      float v = fmaxf(accs[j][r] + bcol, 0.f);
      out[(size_t)(rbase + r) * N + col] = f2bf(v);
    }
  }
}

// ---------------- simple WMMA GEMM (cls head, N=96 pad, f32 logits) -------
__global__ void wmma_gemm_k(const unsigned short* __restrict__ A,
                            const unsigned short* __restrict__ Bp,
                            const float* __restrict__ bias,
                            void* __restrict__ out,
                            int M, int K, int N,
                            int relu, int out_bf16, int ncols) {
  int lane = threadIdx.x & 31;
  int t = blockIdx.x * (blockDim.x >> 5) + (threadIdx.x >> 5);
  int Ntl = N >> 4;
  int Mtl = M >> 4;
  if (t >= Mtl * Ntl) return;            // wave-uniform
  int mt = t / Ntl, nt = t % Ntl;
  int KT = K >> 5;

  int row = mt * 16 + (lane & 15);
  int ko  = (lane >> 4) << 3;
  const unsigned short* arow = A + (size_t)row * K + ko;
  const unsigned short* bptr = Bp + ((size_t)nt * KT * 32 + lane) * 16;

  v8f acc = {};
  for (int kt = 0; kt < KT; ++kt) {
    ABu a;
    a.h[0] = *(const v8bf*)(arow);
    a.h[1] = *(const v8bf*)(arow + 16);
    v16bf bmat = *(const v16bf*)(bptr);
    __builtin_prefetch(bptr + 512, 0, 0);
    acc = __builtin_amdgcn_wmma_f32_16x16x32_bf16(
        false, a.v, false, bmat, (short)0, acc, false, false);
    arow += 32;
    bptr += 512;
  }

  int rbase = mt * 16 + ((lane >> 4) << 3);
  int col   = nt * 16 + (lane & 15);
  if (out_bf16) {
    unsigned short* O = (unsigned short*)out;
    float bcol = bias[col];
    for (int r = 0; r < 8; ++r) {
      float v = acc[r] + bcol;
      if (relu) v = fmaxf(v, 0.f);
      O[(size_t)(rbase + r) * N + col] = f2bf(v);
    }
  } else {
    float* O = (float*)out;
    if (col < ncols) {
      float bcol = bias[col];
      for (int r = 0; r < 8; ++r) {
        float v = acc[r] + bcol;
        if (relu) v = fmaxf(v, 0.f);
        O[(size_t)(rbase + r) * ncols + col] = v;
      }
    }
  }
}

// ---------------- bbox head: deltas = x @ W(1024x4) + b --------------------
__global__ void bbox_k(const unsigned short* __restrict__ x,
                       const float* __restrict__ W,
                       const float* __restrict__ bias,
                       float* __restrict__ deltas, int rows) {
  int idx = blockIdx.x * blockDim.x + threadIdx.x;
  if (idx >= rows * 4) return;
  int r = idx >> 2, j = idx & 3;
  const unsigned short* xr = x + (size_t)r * HID;
  float acc = bias[j];
  for (int k = 0; k < HID; ++k) acc += bf2f(xr[k]) * W[k * 4 + j];
  deltas[idx] = acc;
}

// ---------------- decode + clip (in place on cur) -------------------------
__global__ void decode_k(float* __restrict__ cur,
                         const float* __restrict__ deltas,
                         float wx, float wy, float ww, float wh,
                         const int* __restrict__ img_ptr, int total) {
  int i = blockIdx.x * blockDim.x + threadIdx.x;
  if (i >= total) return;
  float img = (float)(*img_ptr);
  float px1 = cur[i*4+0], py1 = cur[i*4+1], px2 = cur[i*4+2], py2 = cur[i*4+3];
  float pw = fmaxf(px2 - px1, 1e-6f), ph = fmaxf(py2 - py1, 1e-6f);
  float pcx = px1 + 0.5f * pw, pcy = py1 + 0.5f * ph;
  float dx = deltas[i*4+0] * wx, dy = deltas[i*4+1] * wy;
  float dw = fminf(deltas[i*4+2] * ww, CLIP_VAL);
  float dh = fminf(deltas[i*4+3] * wh, CLIP_VAL);
  float cx = dx * pw + pcx, cy = dy * ph + pcy;
  float w = __expf(dw) * pw, h = __expf(dh) * ph;
  cur[i*4+0] = fminf(fmaxf(cx - 0.5f * w, 0.f), img);
  cur[i*4+1] = fminf(fmaxf(cy - 0.5f * h, 0.f), img);
  cur[i*4+2] = fminf(fmaxf(cx + 0.5f * w, 0.f), img);
  cur[i*4+3] = fminf(fmaxf(cy + 0.5f * h, 0.f), img);
}

// ---------------- softmax over 81 classes, one thread per row -------------
__global__ void softmax_k(const float* __restrict__ logits,
                          float* __restrict__ probs, int rows) {
  int r = blockIdx.x * blockDim.x + threadIdx.x;
  if (r >= rows) return;
  const float* L = logits + (size_t)r * NCLS;
  float* P = probs + (size_t)r * NCLS;
  float m = L[0];
  for (int c = 1; c < NCLS; ++c) m = fmaxf(m, L[c]);
  float sum = 0.f;
  for (int c = 0; c < NCLS; ++c) { float e = __expf(L[c] - m); P[c] = e; sum += e; }
  float inv = 1.f / sum;
  for (int c = 0; c < NCLS; ++c) P[c] *= inv;
}

// ---------------- NMS, one block per image --------------------------------
__global__ void nms_k(const float* __restrict__ cur,
                      const float* __restrict__ probs,
                      float* __restrict__ outB, float* __restrict__ outS,
                      float* __restrict__ outL,
                      float* __restrict__ sbuf, float* __restrict__ obbuf,
                      float* __restrict__ areabuf,
                      const int* __restrict__ img_ptr) {
  const int Cm = NCLS - 1;               // 80
  const int M  = NPROP * Cm;             // 40960
  int b   = blockIdx.x;
  int tid = threadIdx.x;
  int nt  = blockDim.x;
  float img = (float)(*img_ptr);
  float* s    = sbuf    + (size_t)b * M;
  float* ob   = obbuf   + (size_t)b * M * 4;
  float* area = areabuf + (size_t)b * M;
  const float* bx = cur   + (size_t)b * NPROP * 4;
  const float* pr = probs + (size_t)b * NPROP * NCLS;

  for (int j = tid; j < M; j += nt) {
    int n = j / Cm, c = j % Cm;
    float sc = pr[n * NCLS + c + 1];
    float x1 = bx[n*4+0], y1 = bx[n*4+1], x2 = bx[n*4+2], y2 = bx[n*4+3];
    bool valid = (sc > 0.05f) && ((x2 - x1) >= 1.0f) && ((y2 - y1) >= 1.0f);
    float off = (float)(c + 1) * (img + 2.0f);
    float o0 = x1 + off, o1 = y1 + off, o2 = x2 + off, o3 = y2 + off;
    ob[j*4+0] = o0; ob[j*4+1] = o1; ob[j*4+2] = o2; ob[j*4+3] = o3;
    area[j] = fmaxf(o2 - o0, 0.f) * fmaxf(o3 - o1, 0.f);
    s[j] = valid ? sc : -1.0f;
  }
  __syncthreads();

  __shared__ float rv[256];
  __shared__ int   ri[256];
  for (int det = 0; det < DETS; ++det) {
    float bv = -1e30f; int bi = 0;
    for (int j = tid; j < M; j += nt) {
      float v = s[j];
      if (v > bv) { bv = v; bi = j; }
    }
    rv[tid] = bv; ri[tid] = bi;
    __syncthreads();
    for (int off = nt >> 1; off > 0; off >>= 1) {
      if (tid < off && rv[tid + off] > rv[tid]) {
        rv[tid] = rv[tid + off]; ri[tid] = ri[tid + off];
      }
      __syncthreads();
    }
    int   i  = ri[0];
    float si = rv[0];
    float b0 = ob[i*4+0], b1 = ob[i*4+1], b2 = ob[i*4+2], b3 = ob[i*4+3];
    float ai = area[i];
    for (int j = tid; j < M; j += nt) {
      float ix1 = fmaxf(b0, ob[j*4+0]);
      float iy1 = fmaxf(b1, ob[j*4+1]);
      float ix2 = fminf(b2, ob[j*4+2]);
      float iy2 = fminf(b3, ob[j*4+3]);
      float inter = fmaxf(ix2 - ix1, 0.f) * fmaxf(iy2 - iy1, 0.f);
      float iou = inter / (ai + area[j] - inter + 1e-9f);
      if (iou > 0.5f) s[j] = -1.0f;
    }
    if (tid == 0) {
      s[i] = -1.0f;
      bool flag = si > 0.0f;
      int n = i / Cm, c = i % Cm;
      float* OB = outB + ((size_t)b * DETS + det) * 4;
      OB[0] = flag ? bx[n*4+0] : 0.f;
      OB[1] = flag ? bx[n*4+1] : 0.f;
      OB[2] = flag ? bx[n*4+2] : 0.f;
      OB[3] = flag ? bx[n*4+3] : 0.f;
      outS[b * DETS + det] = flag ? si : 0.f;
      outL[b * DETS + det] = flag ? (float)(c + 1) : 0.f;
    }
    __syncthreads();
  }
}

// ---------------- host ----------------
extern "C" void kernel_launch(void* const* d_in, const int* in_sizes, int n_in,
                              void* d_out, int out_size, void* d_ws, size_t ws_size,
                              hipStream_t stream) {
  const float* features = (const float*)d_in[0];
  const float* proposals= (const float*)d_in[1];
  const float* fc1_w    = (const float*)d_in[2];
  const float* fc1_b    = (const float*)d_in[3];
  const float* fc2_w    = (const float*)d_in[4];
  const float* fc2_b    = (const float*)d_in[5];
  const float* cls_w    = (const float*)d_in[6];
  const float* cls_b    = (const float*)d_in[7];
  const float* bbox_w   = (const float*)d_in[8];
  const float* bbox_b   = (const float*)d_in[9];
  const int*   img_ptr  = (const int*)d_in[10];

  // workspace carve (256B aligned regions)
  char* p = (char*)d_ws;
  auto carve = [&](size_t bytes) { char* r = p; p += (bytes + 255) & ~(size_t)255; return r; };
  unsigned short* Apool = (unsigned short*)carve((size_t)NROWS * IN_DIM * 2);
  unsigned short* W1p   = (unsigned short*)carve((size_t)IN_DIM * HID * 2);
  unsigned short* W2p   = (unsigned short*)carve((size_t)HID * HID * 2);
  unsigned short* Wcp   = (unsigned short*)carve((size_t)HID * NCLS_PAD * 2);
  unsigned short* x1    = (unsigned short*)carve((size_t)NROWS * HID * 2);
  unsigned short* x2    = (unsigned short*)carve((size_t)NROWS * HID * 2);
  float* logits = (float*)carve((size_t)NROWS * NCLS * 4);
  float* deltas = (float*)carve((size_t)NROWS * 4 * 4);
  float* cur    = (float*)carve((size_t)NROWS * 4 * 4);
  float* probs  = (float*)carve((size_t)NROWS * NCLS * 4);
  float* nms_s  = (float*)carve((size_t)NBATCH * NPROP * 80 * 4);
  float* nms_ob = (float*)carve((size_t)NBATCH * NPROP * 80 * 4 * 4);
  float* nms_ar = (float*)carve((size_t)NBATCH * NPROP * 80 * 4);

  float* outB = (float*)d_out;                     // (2,100,4)
  float* outS = outB + NBATCH * DETS * 4;          // (2,100)
  float* outL = outS + NBATCH * DETS;              // (2,100)

  hipMemcpyAsync(cur, proposals, (size_t)NROWS * 4 * sizeof(float),
                 hipMemcpyDeviceToDevice, stream);

  const float stds[3][4] = {{0.1f, 0.1f, 0.2f, 0.2f},
                            {0.05f, 0.05f, 0.1f, 0.1f},
                            {0.033f, 0.033f, 0.067f, 0.067f}};

  for (int st = 0; st < 3; ++st) {
    roi_align_k<<<NROWS, 256, 0, stream>>>(features, cur, Apool);

    // fc1: (1024 x 12544) @ (12544 x 1024), ReLU, bf16 out
    pack_b_k<<<((size_t)IN_DIM * HID + 255) / 256, 256, 0, stream>>>(
        fc1_w + (size_t)st * IN_DIM * HID, W1p, IN_DIM, HID, HID);
    wmma_gemm_tiled_k<<<(NROWS / 64) * (HID / 128), 256, 0, stream>>>(
        Apool, W1p, fc1_b + st * HID, x1, NROWS, IN_DIM, HID);

    // fc2: (1024 x 1024) @ (1024 x 1024), ReLU, bf16 out
    pack_b_k<<<((size_t)HID * HID + 255) / 256, 256, 0, stream>>>(
        fc2_w + (size_t)st * HID * HID, W2p, HID, HID, HID);
    wmma_gemm_tiled_k<<<(NROWS / 64) * (HID / 128), 256, 0, stream>>>(
        x1, W2p, fc2_b + st * HID, x2, NROWS, HID, HID);

    // cls head: (1024 x 1024) @ (1024 x 81 pad 96), f32 logits
    pack_b_k<<<((size_t)HID * NCLS_PAD + 255) / 256, 256, 0, stream>>>(
        cls_w + (size_t)st * HID * NCLS, Wcp, HID, NCLS_PAD, NCLS);
    wmma_gemm_k<<<(64 * (NCLS_PAD / 16)) / 8, 256, 0, stream>>>(
        x2, Wcp, cls_b + st * NCLS, logits, NROWS, HID, NCLS_PAD, 0, 0, NCLS);

    // bbox head + decode
    bbox_k<<<(NROWS * 4 + 255) / 256, 256, 0, stream>>>(
        x2, bbox_w + (size_t)st * HID * 4, bbox_b + st * 4, deltas, NROWS);
    decode_k<<<(NROWS + 255) / 256, 256, 0, stream>>>(
        cur, deltas, stds[st][0], stds[st][1], stds[st][2], stds[st][3],
        img_ptr, NROWS);
  }

  softmax_k<<<(NROWS + 255) / 256, 256, 0, stream>>>(logits, probs, NROWS);
  nms_k<<<NBATCH, 256, 0, stream>>>(cur, probs, outB, outS, outL,
                                    nms_s, nms_ob, nms_ar, img_ptr);
}
